// SchNetWrapper_27127013441818
// MI455X (gfx1250) — matrix operand order repaired and checked by reading, not compiled, not run on hardware
//
#include <hip/hip_runtime.h>
#include <hip/hip_bf16.h>
#include <math.h>

// ---------------------------------------------------------------------------
// SchNet forward, fused for MI455X (gfx1250, wave32, WMMA f16 16x16x32).
// Edge MLP fully fused (no [E,*] intermediates in HBM) -> bandwidth bound on
// the x-gather + agg scatter-atomics (agg = 25.6MB, L2-resident at 192MB).
// 64 edges/block: each B fragment feeds 4 WMMAs; x-rows prefetched early.
// ---------------------------------------------------------------------------

#define N_ATOMS 50000
#define N_EDGES 1600000
#define N_MOL   500
#define HID     128
#define GAUSS   50
#define GPAD    64
#define LAYERS  3
#define EPB     64          // edges per block (N_EDGES % EPB == 0)
#define NPB     16          // nodes per block (N_ATOMS % NPB == 0)
#define LOG2F_C 0.6931471805599453f
#define PI_F    3.14159265358979f

typedef __attribute__((ext_vector_type(16))) _Float16 v16h;
typedef __attribute__((ext_vector_type(8)))  _Float16 v8h;
typedef __attribute__((ext_vector_type(8)))  float    v8f;

__device__ __forceinline__ float sspf(float v) {
    float sp = (v > 20.f) ? v : ((v < -20.f) ? __expf(v) : log1pf(__expf(v)));
    return sp - LOG2F_C;
}

// 16-bit A fragment (16 rows x K-chunk of 32) from LDS, row stride in halves.
// Lanes 0-15: M=lane, K in kc..kc+15; lanes 16-31: K + 16.
__device__ __forceinline__ v16h load_a_lds(const _Float16* s, int M, int stride,
                                           int kc, int lh) {
    const v8h* p1 = (const v8h*)(s + M * stride + kc + 16 * lh);
    const v8h* p2 = (const v8h*)(s + M * stride + kc + 8 + 16 * lh);
    v8h lo = *p1, hi = *p2;
    v16h a;
#pragma unroll
    for (int i = 0; i < 8; ++i) { a[i] = lo[i]; a[i + 8] = hi[i]; }
    return a;
}

// B fragment from f16 weights stored transposed [n][K]: lane holds column n,
// 16 consecutive K values per lane (lanes 0-15: kc.., lanes 16-31: kc+16..).
__device__ __forceinline__ v16h load_b_t(const _Float16* wT, int n, int K,
                                         int kc, int lh) {
    return *(const v16h*)(wT + (size_t)n * K + kc + 16 * lh);
}

__device__ __forceinline__ v8f wmma16(v16h a, v16h b, v8f c) {
    return __builtin_amdgcn_wmma_f32_16x16x32_f16(false, a, false, b,
                                                  (short)0, c, false, false);
}

// f32x8 -> f16x8 pack helper
__device__ __forceinline__ v8h pack8(float4 f0, float4 f1) {
    v8h o;
    o[0] = (_Float16)f0.x; o[1] = (_Float16)f0.y;
    o[2] = (_Float16)f0.z; o[3] = (_Float16)f0.w;
    o[4] = (_Float16)f1.x; o[5] = (_Float16)f1.y;
    o[6] = (_Float16)f1.z; o[7] = (_Float16)f1.w;
    return o;
}

// ---------------------------------------------------------------------------
// utility kernels
// ---------------------------------------------------------------------------
__global__ void zero4_kernel(float4* p, int n4) {
    int i = blockIdx.x * blockDim.x + threadIdx.x;
    if (i < n4) p[i] = make_float4(0.f, 0.f, 0.f, 0.f);
}

// f32 [L][Ks][128] -> f16 transposed+padded [L][128][Kd]
__global__ void cvt_kernel(const float* __restrict__ src, _Float16* __restrict__ dst,
                           int Ks, int Kd, int total) {
    int i = blockIdx.x * blockDim.x + threadIdx.x;
    if (i >= total) return;
    int k = i % Kd; int rem = i / Kd; int n = rem % HID; int l = rem / HID;
    float v = (k < Ks) ? src[((size_t)l * Ks + k) * HID + n] : 0.f;
    dst[i] = (_Float16)v;
}

__global__ void embed_kernel(const int* __restrict__ z, const float* __restrict__ emb,
                             float* __restrict__ h, int total) {
    int i = blockIdx.x * blockDim.x + threadIdx.x;
    if (i >= total) return;
    int a = i >> 7, c = i & 127;
    h[i] = emb[z[a] * HID + c];
}

// ---------------------------------------------------------------------------
// x = h @ lin1_w   (16 nodes/block, 8 waves, one 16-col tile each; exact grid)
// ---------------------------------------------------------------------------
__global__ __launch_bounds__(256) void nodegemm_kernel(
    const float* __restrict__ h, const _Float16* __restrict__ wT,
    float* __restrict__ x) {
    __shared__ __align__(16) _Float16 sH[NPB * HID];
    int tid = threadIdx.x, lane = tid & 31, wv = tid >> 5;
    int lh = lane >> 4, ln = lane & 15;
    int n0 = blockIdx.x * NPB;
    {   // stage 16x128 f32 -> f16, 8 elems/thread, vectorized
        int base = tid * 8; int m = base >> 7; int k = base & 127;
        const float4* p = (const float4*)(h + (size_t)(n0 + m) * HID + k);
        *(v8h*)(sH + base) = pack8(p[0], p[1]);
    }
    __syncthreads();
    int nIdx = 16 * wv + ln;
    v8f acc = {};
#pragma unroll
    for (int kc = 0; kc < HID; kc += 32) {
        v16h a = load_a_lds(sH, ln, HID, kc, lh);
        v16h b = load_b_t(wT, nIdx, HID, kc, lh);
        acc = wmma16(a, b, acc);
    }
#pragma unroll
    for (int r = 0; r < 8; ++r)
        x[(size_t)(n0 + r + 8 * lh) * HID + nIdx] = acc[r];
}

// ---------------------------------------------------------------------------
// Fused edge kernel: dist -> rbf -> GEMM1 -> ssp -> GEMM2 -> cutoff ->
// gather x[row] -> atomicAdd agg[col].   64 edges/block, 8 waves,
// 4 M-tiles per wave so every B fragment feeds 4 WMMAs.
// ---------------------------------------------------------------------------
__global__ __launch_bounds__(256) void edge_kernel(
    const float* __restrict__ pos, const int* __restrict__ row,
    const int* __restrict__ col, const float* __restrict__ x,
    float* __restrict__ agg,
    const _Float16* __restrict__ w1T, const float* __restrict__ b1,
    const _Float16* __restrict__ w2T, const float* __restrict__ b2) {
    __shared__ __align__(16) _Float16 sRbf[EPB * GPAD];   // 8 KB
    __shared__ __align__(16) _Float16 sAct[EPB * HID];    // 16 KB
    __shared__ float sD[EPB]; __shared__ float sC[EPB];
    __shared__ int sRow[EPB]; __shared__ int sCol[EPB];

    int tid = threadIdx.x, lane = tid & 31, wv = tid >> 5;
    int lh = lane >> 4, ln = lane & 15;
    int e0 = blockIdx.x * EPB;

    if (tid < EPB) {
        int e = e0 + tid;
        int r = row[e], c = col[e];
        float dx = pos[3 * r + 0] - pos[3 * c + 0];
        float dy = pos[3 * r + 1] - pos[3 * c + 1];
        float dz = pos[3 * r + 2] - pos[3 * c + 2];
        float d = sqrtf(dx * dx + dy * dy + dz * dz);
        sRow[tid] = r; sCol[tid] = c; sD[tid] = d;
        sC[tid] = 0.5f * (__cosf(d * (PI_F * 0.1f)) + 1.f);
    }
    __syncthreads();

    // Prefetch the x-rows we will gather later (1 x 128B line per thread,
    // 4 lines x 64 rows) -> global_prefetch, hides gather latency behind GEMMs
    {
        int pe = tid >> 2, ch = tid & 3;
        __builtin_prefetch(x + (size_t)sRow[pe] * HID + ch * 32, 0, 1);
    }

    // Gaussian smearing into LDS (f16), padded 50 -> 64; 16 values/thread
    const float dg = 10.f / 49.f;
    const float coeff = -0.5f / (dg * dg);
    {
        int m = tid >> 2, gb = (tid & 3) * 16;
        float d = sD[m];
        v8h o0, o1;
#pragma unroll
        for (int i = 0; i < 8; ++i) {
            int g = gb + i;
            float t = d - (float)g * dg;
            o0[i] = (_Float16)(g < GAUSS ? __expf(coeff * t * t) : 0.f);
        }
#pragma unroll
        for (int i = 0; i < 8; ++i) {
            int g = gb + 8 + i;
            float t = d - (float)g * dg;
            o1[i] = (_Float16)(g < GAUSS ? __expf(coeff * t * t) : 0.f);
        }
        *(v8h*)(sRbf + m * GPAD + gb) = o0;
        *(v8h*)(sRbf + m * GPAD + gb + 8) = o1;
    }
    __syncthreads();

    int nIdx = 16 * wv + ln;

    // GEMM1: 4 x [16x64] @ [64x128] ; B loaded once per kc, reused 4x
    v8f acc[4] = {};
#pragma unroll
    for (int kc = 0; kc < GPAD; kc += 32) {
        v16h b = load_b_t(w1T, nIdx, GPAD, kc, lh);
#pragma unroll
        for (int mt = 0; mt < 4; ++mt) {
            v16h a = load_a_lds(sRbf + mt * 16 * GPAD, ln, GPAD, kc, lh);
            acc[mt] = wmma16(a, b, acc[mt]);
        }
    }
    float bias1 = b1[nIdx];
#pragma unroll
    for (int mt = 0; mt < 4; ++mt)
#pragma unroll
        for (int r = 0; r < 8; ++r) {
            int m = mt * 16 + r + 8 * lh;
            sAct[m * HID + nIdx] = (_Float16)sspf(acc[mt][r] + bias1);
        }
    __syncthreads();

    // GEMM2: 4 x [16x128] @ [128x128]
    v8f acc2[4] = {};
#pragma unroll
    for (int kc = 0; kc < HID; kc += 32) {
        v16h b = load_b_t(w2T, nIdx, HID, kc, lh);
#pragma unroll
        for (int mt = 0; mt < 4; ++mt) {
            v16h a = load_a_lds(sAct + mt * 16 * HID, ln, HID, kc, lh);
            acc2[mt] = wmma16(a, b, acc2[mt]);
        }
    }
    float bias2 = b2[nIdx];
#pragma unroll
    for (int mt = 0; mt < 4; ++mt)
#pragma unroll
        for (int r = 0; r < 8; ++r) {
            int m = mt * 16 + r + 8 * lh;
            float W = (acc2[mt][r] + bias2) * sC[m];
            float xv = x[(size_t)sRow[m] * HID + nIdx];
            atomicAdd(&agg[(size_t)sCol[m] * HID + nIdx], xv * W);
        }
}

// ---------------------------------------------------------------------------
// Node update: h += (ssp(agg @ lin2 + b)) @ int_w + b_int   (exact grid)
// ---------------------------------------------------------------------------
__global__ __launch_bounds__(256) void nodeupd_kernel(
    const float* __restrict__ agg,
    const _Float16* __restrict__ w2T, const float* __restrict__ b2,
    const _Float16* __restrict__ wiT, const float* __restrict__ bi,
    float* __restrict__ h) {
    __shared__ __align__(16) _Float16 sA[NPB * HID];
    __shared__ __align__(16) _Float16 sAct[NPB * HID];
    int tid = threadIdx.x, lane = tid & 31, wv = tid >> 5;
    int lh = lane >> 4, ln = lane & 15;
    int n0 = blockIdx.x * NPB;
    {
        int base = tid * 8; int m = base >> 7; int k = base & 127;
        const float4* p = (const float4*)(agg + (size_t)(n0 + m) * HID + k);
        *(v8h*)(sA + base) = pack8(p[0], p[1]);
    }
    __syncthreads();

    int nIdx = 16 * wv + ln;
    v8f acc = {};
#pragma unroll
    for (int kc = 0; kc < HID; kc += 32) {
        v16h a = load_a_lds(sA, ln, HID, kc, lh);
        v16h b = load_b_t(w2T, nIdx, HID, kc, lh);
        acc = wmma16(a, b, acc);
    }
    float bias = b2[nIdx];
#pragma unroll
    for (int r = 0; r < 8; ++r) {
        int m = r + 8 * lh;
        sAct[m * HID + nIdx] = (_Float16)sspf(acc[r] + bias);
    }
    __syncthreads();

    v8f acc2 = {};
#pragma unroll
    for (int kc = 0; kc < HID; kc += 32) {
        v16h a = load_a_lds(sAct, ln, HID, kc, lh);
        v16h b = load_b_t(wiT, nIdx, HID, kc, lh);
        acc2 = wmma16(a, b, acc2);
    }
    float biasi = bi[nIdx];
#pragma unroll
    for (int r = 0; r < 8; ++r) {
        size_t idx = (size_t)(n0 + r + 8 * lh) * HID + nIdx;
        h[idx] = h[idx] + acc2[r] + biasi;   // residual
    }
}

// ---------------------------------------------------------------------------
// Head: per-atom  e = ssp(h@out1+b1) @ out2 + b2;  atomicAdd to out[batch]
// ---------------------------------------------------------------------------
__global__ __launch_bounds__(64) void head_kernel(
    const float* __restrict__ h, const float* __restrict__ w1,
    const float* __restrict__ b1, const float* __restrict__ w2,
    const float* __restrict__ b2, const int* __restrict__ batch,
    float* __restrict__ out, int nN) {
    __shared__ float sH[64 * 129];   // pad to 129 -> conflict-free
    int t = threadIdx.x; int a0 = blockIdx.x * 64;
    for (int v = t; v < 64 * HID; v += 64) {
        int a = v >> 7, k = v & 127; int node = a0 + a;
        sH[a * 129 + k] = (node < nN) ? h[(size_t)node * HID + k] : 0.f;
    }
    __syncthreads();
    float e = 0.f;
    for (int j = 0; j < 64; ++j) {
        float acc = b1[j];
#pragma unroll 8
        for (int k = 0; k < HID; ++k) acc += sH[t * 129 + k] * w1[k * 64 + j];
        e += sspf(acc) * w2[j];
    }
    e += b2[0];
    int node = a0 + t;
    if (node < nN) atomicAdd(&out[batch[node]], e);
}

// ---------------------------------------------------------------------------
extern "C" void kernel_launch(void* const* d_in, const int* in_sizes, int n_in,
                              void* d_out, int out_size, void* d_ws, size_t ws_size,
                              hipStream_t stream) {
    const int*   z      = (const int*)  d_in[0];
    const float* pos    = (const float*)d_in[1];
    const int*   batch  = (const int*)  d_in[2];
    const int*   eidx   = (const int*)  d_in[3];
    const int*   row    = eidx;
    const int*   col    = eidx + N_EDGES;
    const float* emb    = (const float*)d_in[4];
    const float* mlp1_w = (const float*)d_in[5];
    const float* mlp1_b = (const float*)d_in[6];
    const float* mlp2_w = (const float*)d_in[7];
    const float* mlp2_b = (const float*)d_in[8];
    const float* lin1_w = (const float*)d_in[9];
    const float* lin2_w = (const float*)d_in[10];
    const float* lin2_b = (const float*)d_in[11];
    const float* int_w  = (const float*)d_in[12];
    const float* int_b  = (const float*)d_in[13];
    const float* out1_w = (const float*)d_in[14];
    const float* out1_b = (const float*)d_in[15];
    const float* out2_w = (const float*)d_in[16];
    const float* out2_b = (const float*)d_in[17];
    float* out = (float*)d_out;

    // workspace layout (bytes): h | x | agg | f16 weights (~77.3 MB total)
    char* ws = (char*)d_ws;
    const size_t NF = (size_t)N_ATOMS * HID;           // 6,400,000 floats
    float* h   = (float*)(ws);
    float* x   = (float*)(ws + NF * 4);
    float* agg = (float*)(ws + NF * 8);
    _Float16* w1T = (_Float16*)(ws + NF * 12);         // [3][128][64]
    _Float16* w2T = w1T + (size_t)LAYERS * HID * GPAD; // [3][128][128]
    _Float16* l1T = w2T + (size_t)LAYERS * HID * HID;
    _Float16* l2T = l1T + (size_t)LAYERS * HID * HID;
    _Float16* inT = l2T + (size_t)LAYERS * HID * HID;

    // zero output (harness poisons it); N_MOL=500 -> 125 float4
    zero4_kernel<<<1, 256, 0, stream>>>((float4*)out, N_MOL / 4);

    // convert weights to f16, transposed [l][n][k]
    {
        int t1 = LAYERS * HID * GPAD;
        int t2 = LAYERS * HID * HID;
        cvt_kernel<<<(t1 + 255) / 256, 256, 0, stream>>>(mlp1_w, w1T, GAUSS, GPAD, t1);
        cvt_kernel<<<(t2 + 255) / 256, 256, 0, stream>>>(mlp2_w, w2T, HID, HID, t2);
        cvt_kernel<<<(t2 + 255) / 256, 256, 0, stream>>>(lin1_w, l1T, HID, HID, t2);
        cvt_kernel<<<(t2 + 255) / 256, 256, 0, stream>>>(lin2_w, l2T, HID, HID, t2);
        cvt_kernel<<<(t2 + 255) / 256, 256, 0, stream>>>(int_w,  inT, HID, HID, t2);
    }

    // h = emb[z]
    embed_kernel<<<(int)((NF + 255) / 256), 256, 0, stream>>>(z, emb, h, (int)NF);

    const int nodeBlocks = N_ATOMS / NPB;      // 3125 (exact)
    const int edgeBlocks = N_EDGES / EPB;      // 25000 (exact)
    for (int l = 0; l < LAYERS; ++l) {
        nodegemm_kernel<<<nodeBlocks, 256, 0, stream>>>(
            h, l1T + (size_t)l * HID * HID, x);
        zero4_kernel<<<(int)((NF / 4 + 255) / 256), 256, 0, stream>>>(
            (float4*)agg, (int)(NF / 4));
        edge_kernel<<<edgeBlocks, 256, 0, stream>>>(
            pos, row, col, x, agg,
            w1T + (size_t)l * HID * GPAD, mlp1_b + l * HID,
            w2T + (size_t)l * HID * HID,  mlp2_b + l * HID);
        nodeupd_kernel<<<nodeBlocks, 256, 0, stream>>>(
            agg, l2T + (size_t)l * HID * HID, lin2_b + l * HID,
            inT + (size_t)l * HID * HID, int_b + l * HID, h);
    }

    head_kernel<<<(N_ATOMS + 63) / 64, 64, 0, stream>>>(
        h, out1_w, out1_b, out2_w, out2_b, batch, out, N_ATOMS);
}
